// Silco_22411139350740
// MI455X (gfx1250) — matrix operand order repaired and checked by reading, not compiled, not run on hardware
//
#include <hip/hip_runtime.h>
#include <hip/hip_bf16.h>

// ---------------------------------------------------------------------------
// GraphNet forward on gfx1250 (MI455X): fused pairwise-MLP via v_wmma_f32_16x16x32_f16
// M=32 register blocking per wave: A fragments cached in VGPRs across the whole
// layer, each B fragment shared by two WMMAs.
// ---------------------------------------------------------------------------

typedef __attribute__((ext_vector_type(16))) _Float16 v16h;
typedef __attribute__((ext_vector_type(8)))  float    v8f;

#define BATCH 8
#define NPTS  256

union Frag {
    v16h h;
    int4 q[2];
};

// One dense layer for one wave's 32-row tile:
//   actOut[32][Nd] = leaky( actIn[32][K] @ W[K][Nd] + bias ),
// weights stored transposed+padded as wt[Nd][K] (f16), K = KC*32.
template <int KC>
__device__ __forceinline__ void mlp_layer_wave32(
    const _Float16* __restrict__ actIn, int inStride,
    _Float16* __restrict__ actOut, int outStride,
    const _Float16* __restrict__ wt,
    const float* __restrict__ bias,
    int Nd, int lane)
{
    const int rlo = lane & 15;   // A row (within 16-row sub-tile) / B col / C col
    const int khi = lane >> 4;   // K-half selector
    constexpr int K = KC * 32;

    // Cache all A fragments (2 M-sub-tiles x KC chunks) in registers.
    Frag a[2][KC];
#pragma unroll
    for (int m = 0; m < 2; ++m) {
#pragma unroll
        for (int kc = 0; kc < KC; ++kc) {
            const _Float16* pa = actIn + (m * 16 + rlo) * inStride + kc * 32 + 8 * khi;
            a[m][kc].q[0] = *(const int4*)(pa);
            a[m][kc].q[1] = *(const int4*)(pa + 16);
        }
    }

#pragma unroll 1
    for (int n0 = 0; n0 < Nd; n0 += 16) {
        const float bv = bias[n0 + rlo];
        v8f c0, c1;
#pragma unroll
        for (int r = 0; r < 8; ++r) { c0[r] = bv; c1[r] = bv; }

#pragma unroll
        for (int kc = 0; kc < KC; ++kc) {
            Frag b;
            // B: 32x16 f16, lane rlo = col; elem e<8 -> K = kc*32 + 8*khi + e, e>=8 -> K = kc*32 + 16 + 8*khi + (e-8)
            const _Float16* pb = wt + (size_t)(n0 + rlo) * K + kc * 32 + 8 * khi;
            b.q[0] = *(const int4*)(pb);
            b.q[1] = *(const int4*)(pb + 16);
            c0 = __builtin_amdgcn_wmma_f32_16x16x32_f16(
                     false, a[0][kc].h, false, b.h, (short)0, c0, false, false);
            c1 = __builtin_amdgcn_wmma_f32_16x16x32_f16(
                     false, a[1][kc].h, false, b.h, (short)0, c1, false, false);
        }

        // C/D: VGPR r -> row r + 8*khi (+16 for second sub-tile), col n0 + rlo
#pragma unroll
        for (int r = 0; r < 8; ++r) {
            float v0 = c0[r];
            v0 = (v0 >= 0.0f) ? v0 : 0.01f * v0;
            actOut[(r + 8 * khi) * outStride + n0 + rlo] = (_Float16)v0;
            float v1 = c1[r];
            v1 = (v1 >= 0.0f) ? v1 : 0.01f * v1;
            actOut[(16 + r + 8 * khi) * outStride + n0 + rlo] = (_Float16)v1;
        }
    }
}

// Fused adjacency MLP: per 32-row tile (fixed b,i; j0..j0+31), compute
// h = |x_i - x_j| then the 5-layer chain, write logits.
template <int KP0>
__global__ __launch_bounds__(64) void adj_kernel(
    const _Float16* __restrict__ xh,     // [B*N*KP0] f16, zero-padded features
    const _Float16* __restrict__ wt0,    // [192][KP0]
    const _Float16* __restrict__ wt1,    // [192][192]
    const _Float16* __restrict__ wt2,    // [192][192]
    const _Float16* __restrict__ wt3,    // [96][192]
    const float* __restrict__ bias0, const float* __restrict__ bias1,
    const float* __restrict__ bias2, const float* __restrict__ bias3,
    const float* __restrict__ w4,        // [96] f32
    const float* __restrict__ b4,        // [1]  f32
    float* __restrict__ logits,          // [B*N*N]
    int nWaves)
{
    __shared__ _Float16 smem[2][2][32 * 192];   // per-wave ping-pong activation tiles

    const int lane  = threadIdx.x & 31;
    const int wslot = threadIdx.x >> 5;
    _Float16* bufA = smem[wslot][0];
    _Float16* bufB = smem[wslot][1];
    constexpr int KC0 = KP0 / 32;

    const int tilesTotal = (BATCH * NPTS * NPTS) / 32;
    const int waveId = blockIdx.x * 2 + wslot;

    for (int t = waveId; t < tilesTotal; t += nWaves) {
        const int b   = t >> 11;           // 2048 tiles per batch
        const int rem = t & 2047;
        const int i   = rem >> 3;
        const int j0  = (rem & 7) << 5;

        // WAR fence vs previous iteration's LDS reads
        asm volatile("s_wait_dscnt 0" ::: "memory");

        // act0[32][KP0] = |x_i - x_j|, one row per lane (padding: 0-0 = 0)
        {
            const _Float16* xi = xh + (size_t)(b * NPTS + i) * KP0;
            const _Float16* xj = xh + (size_t)(b * NPTS + j0 + lane) * KP0;
            for (int k = 0; k < KP0; ++k) {
                float d = (float)xi[k] - (float)xj[k];
                bufA[lane * KP0 + k] = (_Float16)(d < 0.0f ? -d : d);
            }
        }
        asm volatile("s_wait_dscnt 0" ::: "memory");

        mlp_layer_wave32<KC0>(bufA, KP0, bufB, 192, wt0, bias0, 192, lane);
        asm volatile("s_wait_dscnt 0" ::: "memory");
        mlp_layer_wave32<6>(bufB, 192, bufA, 192, wt1, bias1, 192, lane);
        asm volatile("s_wait_dscnt 0" ::: "memory");
        mlp_layer_wave32<6>(bufA, 192, bufB, 192, wt2, bias2, 192, lane);
        asm volatile("s_wait_dscnt 0" ::: "memory");
        mlp_layer_wave32<6>(bufB, 192, bufA, 192, wt3, bias3, 96, lane);
        asm volatile("s_wait_dscnt 0" ::: "memory");

        // final 96 -> 1 dot, one row per lane (w4 indices lane-uniform -> s_loads)
        {
            union { int4 q; _Float16 h[8]; } u;
            float acc = b4[0];
            const _Float16* pr = bufA + lane * 192;
#pragma unroll
            for (int c = 0; c < 12; ++c) {
                u.q = *(const int4*)(pr + c * 8);
#pragma unroll
                for (int e = 0; e < 8; ++e)
                    acc += (float)u.h[e] * w4[c * 8 + e];
            }
            logits[(size_t)(b * NPTS + i) * NPTS + j0 + lane] = acc;
        }
    }
}

// In-place masked softmax over j for one (b,i) row of 256 logits.
__global__ void softmax_kernel(float* __restrict__ a)
{
    __shared__ float red[256];
    const int row = blockIdx.x;          // b*N + i
    const int i   = row & 255;
    const int tid = threadIdx.x;

    float v = a[(size_t)row * 256 + tid];
    if (tid == i) v -= 1e8f;

    red[tid] = v;
    __syncthreads();
    for (int s = 128; s > 0; s >>= 1) {
        if (tid < s) red[tid] = fmaxf(red[tid], red[tid + s]);
        __syncthreads();
    }
    const float m = red[0];
    __syncthreads();

    const float e = __expf(v - m);
    red[tid] = e;
    __syncthreads();
    for (int s = 128; s > 0; s >>= 1) {
        if (tid < s) red[tid] += red[tid + s];
        __syncthreads();
    }
    a[(size_t)row * 256 + tid] = e / red[0];
}

// x1 = leaky([x, A0 x] @ g0_w^T + g0_b); xc = [x, x1]; also f16-padded xch.
__global__ void g0_kernel(
    const float* __restrict__ x,      // [B*N*24]
    const float* __restrict__ a0,     // [B*N*N]
    const float* __restrict__ w,      // [48*48] (out,in)
    const float* __restrict__ bias,   // [48]
    float* __restrict__ xc,           // [B*N*72]
    _Float16* __restrict__ xch)       // [B*N*96] padded
{
    __shared__ float arow[256];
    __shared__ float agg[48];
    __shared__ float xcs[72];

    const int row = blockIdx.x;       // b*N + i
    const int b   = row >> 8;
    const int tid = threadIdx.x;

    arow[tid] = a0[(size_t)row * 256 + tid];
    __syncthreads();

    if (tid < 24) {
        float s = 0.0f;
        for (int j = 0; j < 256; ++j)
            s += arow[j] * x[(size_t)(b * 256 + j) * 24 + tid];
        agg[24 + tid] = s;
        agg[tid] = x[(size_t)row * 24 + tid];
    }
    __syncthreads();

    float x1 = 0.0f;
    if (tid < 48) {
        float s = bias[tid];
        for (int c = 0; c < 48; ++c) s += agg[c] * w[tid * 48 + c];
        x1 = (s >= 0.0f) ? s : 0.01f * s;
    }
    if (tid < 24) xcs[tid] = agg[tid];
    if (tid < 48) xcs[24 + tid] = x1;
    __syncthreads();

    if (tid < 72) xc[(size_t)row * 72 + tid] = xcs[tid];
    if (tid < 96) xch[(size_t)row * 96 + tid] = (_Float16)(tid < 72 ? xcs[tid] : 0.0f);
}

// out = [xc, A1 xc] @ g1_w^T + g1_b  -> scalar per (b,i)
__global__ void g1_kernel(
    const float* __restrict__ xc,     // [B*N*72]
    const float* __restrict__ a1,     // [B*N*N]
    const float* __restrict__ w,      // [144]
    const float* __restrict__ bias,   // [1]
    float* __restrict__ out)          // [B*N]
{
    __shared__ float arow[256];
    __shared__ float part[144];

    const int row = blockIdx.x;
    const int b   = row >> 8;
    const int tid = threadIdx.x;

    arow[tid] = a1[(size_t)row * 256 + tid];
    __syncthreads();

    if (tid < 72) {
        float s = 0.0f;
        for (int j = 0; j < 256; ++j)
            s += arow[j] * xc[(size_t)(b * 256 + j) * 72 + tid];
        part[72 + tid] = s * w[72 + tid];
        part[tid] = xc[(size_t)row * 72 + tid] * w[tid];
    }
    __syncthreads();

    if (tid == 0) {
        float s = bias[0];
        for (int c = 0; c < 144; ++c) s += part[c];
        out[row] = s;
    }
}

// Convert weights [Korig][Nd] f32 -> transposed, K-padded f16 [Nd][KP]
__global__ void wconv_kernel(const float* __restrict__ w, _Float16* __restrict__ wt,
                             int Korig, int KP, int Nd)
{
    const int idx = blockIdx.x * 256 + threadIdx.x;
    if (idx >= KP * Nd) return;
    const int n = idx / KP;
    const int k = idx % KP;
    wt[idx] = (_Float16)((k < Korig) ? w[k * Nd + n] : 0.0f);
}

// x [B*N][24] f32 -> xh [B*N][32] f16 zero-padded
__global__ void xconv_kernel(const float* __restrict__ x, _Float16* __restrict__ xh)
{
    const int idx = blockIdx.x * 256 + threadIdx.x;   // B*N*32 = 65536
    if (idx >= BATCH * NPTS * 32) return;
    const int r = idx >> 5;
    const int k = idx & 31;
    xh[idx] = (_Float16)((k < 24) ? x[r * 24 + k] : 0.0f);
}

extern "C" void kernel_launch(void* const* d_in, const int* in_sizes, int n_in,
                              void* d_out, int out_size, void* d_ws, size_t ws_size,
                              hipStream_t stream)
{
    (void)in_sizes; (void)n_in; (void)out_size; (void)ws_size;

    const float* x = (const float*)d_in[0];
    const float* a0w[5] = {(const float*)d_in[1], (const float*)d_in[3],
                           (const float*)d_in[5], (const float*)d_in[7],
                           (const float*)d_in[9]};
    const float* a0b[5] = {(const float*)d_in[2], (const float*)d_in[4],
                           (const float*)d_in[6], (const float*)d_in[8],
                           (const float*)d_in[10]};
    const float* a1w[5] = {(const float*)d_in[11], (const float*)d_in[13],
                           (const float*)d_in[15], (const float*)d_in[17],
                           (const float*)d_in[19]};
    const float* a1b[5] = {(const float*)d_in[12], (const float*)d_in[14],
                           (const float*)d_in[16], (const float*)d_in[18],
                           (const float*)d_in[20]};
    const float* g0w = (const float*)d_in[21];
    const float* g0b = (const float*)d_in[22];
    const float* g1w = (const float*)d_in[23];
    const float* g1b = (const float*)d_in[24];

    char* ws = (char*)d_ws;
    size_t off = 0;
    auto take = [&](size_t bytes) -> char* {
        char* p = ws + off;
        off = (off + bytes + 255) & ~(size_t)255;
        return p;
    };

    _Float16* wt00 = (_Float16*)take(192 * 32 * 2);
    _Float16* wt01 = (_Float16*)take(192 * 192 * 2);
    _Float16* wt02 = (_Float16*)take(192 * 192 * 2);
    _Float16* wt03 = (_Float16*)take(96 * 192 * 2);
    _Float16* wt10 = (_Float16*)take(192 * 96 * 2);
    _Float16* wt11 = (_Float16*)take(192 * 192 * 2);
    _Float16* wt12 = (_Float16*)take(192 * 192 * 2);
    _Float16* wt13 = (_Float16*)take(96 * 192 * 2);
    _Float16* xh   = (_Float16*)take((size_t)BATCH * NPTS * 32 * 2);
    _Float16* xch  = (_Float16*)take((size_t)BATCH * NPTS * 96 * 2);
    float*    xc   = (float*)take((size_t)BATCH * NPTS * 72 * 4);

    float* out_ptr = (float*)d_out;                       // [B*N]
    float* a0_ptr  = out_ptr + BATCH * NPTS;              // [B*N*N]
    float* a1_ptr  = a0_ptr + (size_t)BATCH * NPTS * NPTS;

    // ---- precision prep: f16 transposed/padded weights + f16 padded x ----
    wconv_kernel<<<(192 * 32 + 255) / 256, 256, 0, stream>>>(a0w[0], wt00, 24, 32, 192);
    wconv_kernel<<<(192 * 192 + 255) / 256, 256, 0, stream>>>(a0w[1], wt01, 192, 192, 192);
    wconv_kernel<<<(192 * 192 + 255) / 256, 256, 0, stream>>>(a0w[2], wt02, 192, 192, 192);
    wconv_kernel<<<(96 * 192 + 255) / 256, 256, 0, stream>>>(a0w[3], wt03, 192, 192, 96);
    wconv_kernel<<<(192 * 96 + 255) / 256, 256, 0, stream>>>(a1w[0], wt10, 72, 96, 192);
    wconv_kernel<<<(192 * 192 + 255) / 256, 256, 0, stream>>>(a1w[1], wt11, 192, 192, 192);
    wconv_kernel<<<(192 * 192 + 255) / 256, 256, 0, stream>>>(a1w[2], wt12, 192, 192, 192);
    wconv_kernel<<<(96 * 192 + 255) / 256, 256, 0, stream>>>(a1w[3], wt13, 192, 192, 96);
    xconv_kernel<<<(BATCH * NPTS * 32 + 255) / 256, 256, 0, stream>>>(x, xh);

    // ---- adjacency 0: logits -> softmax (in place in d_out a0 region) ----
    adj_kernel<32><<<512, 64, 0, stream>>>(xh, wt00, wt01, wt02, wt03,
                                           a0b[0], a0b[1], a0b[2], a0b[3],
                                           a0w[4], a0b[4], a0_ptr, 512 * 2);
    softmax_kernel<<<BATCH * NPTS, 256, 0, stream>>>(a0_ptr);

    // ---- graph module 0 -> xc (f32) and xch (f16 padded) ----
    g0_kernel<<<BATCH * NPTS, 256, 0, stream>>>(x, a0_ptr, g0w, g0b, xc, xch);

    // ---- adjacency 1 ----
    adj_kernel<96><<<512, 64, 0, stream>>>(xch, wt10, wt11, wt12, wt13,
                                           a1b[0], a1b[1], a1b[2], a1b[3],
                                           a1w[4], a1b[4], a1_ptr, 512 * 2);
    softmax_kernel<<<BATCH * NPTS, 256, 0, stream>>>(a1_ptr);

    // ---- final graph module -> out ----
    g1_kernel<<<BATCH * NPTS, 256, 0, stream>>>(xc, a1_ptr, g1w, g1b, out_ptr);
}
